// ModulatedMambaBlock_11725260718317
// MI455X (gfx1250) — matrix operand order repaired and checked by reading, not compiled, not run on hardware
//
#include <hip/hip_runtime.h>
#include <hip/hip_bf16.h>
#include <math.h>

#define D_MODEL 768
#define D_COND 128
#define D_STATE 64
#define D_CONV 4
#define D_INNER 1536
#define DT_RANK 48
#define DT_PAD 64          // DT_RANK zero-padded to a multiple of 32
#define BATCH 4
#define SEQ 2048
#define NROWS (BATCH * SEQ)   // 8192

typedef __attribute__((ext_vector_type(16))) _Float16 v16h;
typedef __attribute__((ext_vector_type(8)))  float    v8f;

// ---------------------------------------------------------------------------
// f32 -> f16 bulk convert
// ---------------------------------------------------------------------------
__global__ void cvt_f16_kernel(const float* __restrict__ src,
                               _Float16* __restrict__ dst, int n) {
  const int i = blockIdx.x * blockDim.x + threadIdx.x;
  if (i < n) dst[i] = (_Float16)src[i];
}

// ---------------------------------------------------------------------------
// Row-wise pack + zero-pad: dst[r, 0:kvalid] = src[r, 0:kvalid], rest = 0.
// Used to pad the K=48 dt-rank operands to K=64.
// ---------------------------------------------------------------------------
__global__ void pack_pad_f16_kernel(const float* __restrict__ src, int sld,
                                    int kvalid, _Float16* __restrict__ dst,
                                    int dld, int total) {
  const int i = blockIdx.x * blockDim.x + threadIdx.x;
  if (i >= total) return;
  const int r = i / dld, j = i % dld;
  dst[i] = (_Float16)((j < kvalid) ? src[(long long)r * sld + j] : 0.0f);
}

// ---------------------------------------------------------------------------
// WMMA GEMM: C[M,N] = A[M,K](f16) * W[N,K](f16)^T (+bias) (+softplus)
// 256 threads = 8 waves; each wave computes 16 rows x 128 cols (8 16x16
// tiles, A fragment reused 8x). K multiple of 32, N multiple of 16.
// Out-of-range N tiles are clamped (redundant compute, EXEC stays all-1s);
// stores are guarded. EPI: 0 = bias, 1 = bias + softplus.
// ---------------------------------------------------------------------------
template<int EPI>
__global__ __launch_bounds__(256)
void wmma_gemm_kernel(const _Float16* __restrict__ A, int lda,
                      const _Float16* __restrict__ W, int ldw,
                      const float* __restrict__ bias,
                      float* __restrict__ C, int ldc, int N, int K) {
  const int lane   = threadIdx.x & 31;
  const int wave   = threadIdx.x >> 5;
  const int row0   = blockIdx.x << 4;
  const int nTiles = N >> 4;
  const int tileN0 = (blockIdx.y * 8 + wave) * 8;

  // per-lane fragment base pointers: lane covers row (lane&15), 16 contiguous
  // K elements starting at (lane>>4)*16 (contiguous k order, same for A and B)
  const _Float16* aptr = A + (long long)(row0 + (lane & 15)) * lda + (lane >> 4) * 16;
  const _Float16* bptr[8];
  int tns[8];
#pragma unroll
  for (int t = 0; t < 8; ++t) {
    int tn = tileN0 + t;
    tns[t] = tn;
    tn = (tn < nTiles) ? tn : (nTiles - 1);       // clamp: keep EXEC full
    bptr[t] = W + (long long)((tn << 4) + (lane & 15)) * ldw + (lane >> 4) * 16;
  }

  v8f acc[8] = {};
  for (int k0 = 0; k0 < K; k0 += 32) {
    if (k0 + 32 < K)  // prefetch next A K-slab -> global_prefetch_b8
      __builtin_prefetch(aptr + k0 + 32, 0, 1);
    const v16h a = *(const v16h*)(aptr + k0);
#pragma unroll
    for (int t = 0; t < 8; ++t) {
      const v16h b = *(const v16h*)(bptr[t] + k0);
      acc[t] = __builtin_amdgcn_wmma_f32_16x16x32_f16(
          false, a, false, b, (short)0, acc[t], false, false);
    }
  }

  // C/D layout: VGPR r, lanes 0-15 -> M=r, lanes 16-31 -> M=r+8; N = lane&15
  const int colLane = lane & 15;
  const int rowOff  = (lane >> 4) << 3;
#pragma unroll
  for (int t = 0; t < 8; ++t) {
    if (tns[t] >= nTiles) continue;               // wave-uniform guard
    const int col = (tns[t] << 4) + colLane;
    const float bv = bias ? bias[col] : 0.0f;
#pragma unroll
    for (int r = 0; r < 8; ++r) {
      const int row = row0 + rowOff + r;
      float v = acc[t][r] + bv;
      if (EPI == 1) v = (v > 20.0f) ? v : log1pf(__expf(v));  // softplus
      C[(long long)row * ldc + col] = v;
    }
  }
}

// ---------------------------------------------------------------------------
// Causal depthwise conv (D_CONV=4) + SiLU. xin = xz[..., :D_INNER].
// Emits both f32 (for the scan) and f16 (for the x_proj GEMM).
// ---------------------------------------------------------------------------
__global__ void conv_silu_kernel(const float* __restrict__ xz,
                                 const float* __restrict__ conv_w,
                                 const float* __restrict__ conv_b,
                                 float* __restrict__ xc,
                                 _Float16* __restrict__ xch, int total) {
  const int i = blockIdx.x * blockDim.x + threadIdx.x;
  if (i >= total) return;
  const int d = i % D_INNER;
  const int l = (i / D_INNER) % SEQ;
  const int b = i / (D_INNER * SEQ);
  float s = conv_b[d];
  const float* w = conv_w + d * D_CONV;
#pragma unroll
  for (int t = 0; t < D_CONV; ++t) {
    const int ls = l + t - (D_CONV - 1);
    if (ls >= 0)
      s += xz[((long long)b * SEQ + ls) * (2 * D_INNER) + d] * w[t];
  }
  const float v = s / (1.0f + __expf(-s));  // silu
  xc[i]  = v;
  xch[i] = (_Float16)v;
}

// ---------------------------------------------------------------------------
// Selective scan: one wave32 per (b,d) row; lane owns states n and n+32.
// Fuses D-skip and silu(z) gate; writes ys (f16, feeds out_proj GEMM) and
// the final hidden state (f32, part of d_out).
// ---------------------------------------------------------------------------
__global__ __launch_bounds__(256)
void scan_kernel(const float* __restrict__ xc, const float* __restrict__ dt,
                 const float* __restrict__ dbc, const float* __restrict__ xz,
                 const float* __restrict__ A_log, const float* __restrict__ Dskip,
                 _Float16* __restrict__ ysh, float* __restrict__ new_state) {
  const int wid = blockIdx.x * (blockDim.x >> 5) + (threadIdx.x >> 5);
  if (wid >= BATCH * D_INNER) return;
  const int b = wid / D_INNER;
  const int d = wid % D_INNER;
  const int lane = threadIdx.x & 31;

  const float A0 = -__expf(A_log[(long long)d * D_STATE + lane]);
  const float A1 = -__expf(A_log[(long long)d * D_STATE + lane + 32]);
  const float Dd = Dskip[d];
  float h0 = 0.0f, h1 = 0.0f;

  for (int l = 0; l < SEQ; ++l) {
    const long long row = (long long)b * SEQ + l;
    const float xc_t = xc[row * D_INNER + d];
    const float dt_t = dt[row * D_INNER + d];
    const float* dbc_row = dbc + row * (DT_RANK + 2 * D_STATE);
    const float B0 = dbc_row[DT_RANK + lane];
    const float B1 = dbc_row[DT_RANK + lane + 32];
    const float C0 = dbc_row[DT_RANK + D_STATE + lane];
    const float C1 = dbc_row[DT_RANK + D_STATE + lane + 32];
    const float dtx = dt_t * xc_t;
    h0 = __expf(dt_t * A0) * h0 + dtx * B0;
    h1 = __expf(dt_t * A1) * h1 + dtx * B1;
    float y = h0 * C0 + h1 * C1;
#pragma unroll
    for (int o = 16; o; o >>= 1) y += __shfl_xor(y, o, 32);
    if (lane == 0) {
      const float z  = xz[row * (2 * D_INNER) + D_INNER + d];
      const float yv = (y + Dd * xc_t) * (z / (1.0f + __expf(-z)));
      ysh[row * D_INNER + d] = (_Float16)yv;
    }
  }
  new_state[((long long)b * D_INNER + d) * D_STATE + lane]      = h0;
  new_state[((long long)b * D_INNER + d) * D_STATE + lane + 32] = h1;
}

// ---------------------------------------------------------------------------
// AdaLN projection: ss[b, n] = cond[b,:] . ada_w[n,:] + ada_b[n]  (4 x 1536)
// ---------------------------------------------------------------------------
__global__ void ada_kernel(const float* __restrict__ cond,
                           const float* __restrict__ ada_w,
                           const float* __restrict__ ada_b,
                           float* __restrict__ ss) {
  const int j = blockIdx.x * blockDim.x + threadIdx.x;
  if (j >= BATCH * 2 * D_MODEL) return;
  const int b = j / (2 * D_MODEL);
  const int n = j % (2 * D_MODEL);
  const float* c = cond + b * D_COND;
  const float* w = ada_w + (long long)n * D_COND;
  float s = ada_b[n];
#pragma unroll 4
  for (int k = 0; k < D_COND; ++k) s += c[k] * w[k];
  ss[j] = s;
}

// ---------------------------------------------------------------------------
// Fused: LN(x+mam, w, b) -> LN(.) * (1+scale) + shift. One 256-thread block
// per (b,l) row; shfl + LDS two-level reductions, twice.
// ---------------------------------------------------------------------------
__global__ __launch_bounds__(256)
void final_ln_kernel(const float* __restrict__ x, const float* __restrict__ mam,
                     const float* __restrict__ ln_w, const float* __restrict__ ln_b,
                     const float* __restrict__ ss, float* __restrict__ out) {
  __shared__ float buf[D_MODEL];
  __shared__ float redS[8], redQ[8];
  const int row  = blockIdx.x;
  const int b    = row / SEQ;
  const int tid  = threadIdx.x;
  const int lane = tid & 31;
  const int wave = tid >> 5;
  const float eps = 1e-5f;
  const float* xr = x   + (long long)row * D_MODEL;
  const float* mr = mam + (long long)row * D_MODEL;

  float s = 0.0f, q = 0.0f;
  for (int n = tid; n < D_MODEL; n += 256) {
    const float v = xr[n] + mr[n];
    buf[n] = v; s += v; q += v * v;
  }
#pragma unroll
  for (int o = 16; o; o >>= 1) { s += __shfl_xor(s, o, 32); q += __shfl_xor(q, o, 32); }
  if (lane == 0) { redS[wave] = s; redQ[wave] = q; }
  __syncthreads();
  if (wave == 0) {
    s = (lane < 8) ? redS[lane] : 0.0f;
    q = (lane < 8) ? redQ[lane] : 0.0f;
#pragma unroll
    for (int o = 16; o; o >>= 1) { s += __shfl_xor(s, o, 32); q += __shfl_xor(q, o, 32); }
    if (lane == 0) { redS[0] = s; redQ[0] = q; }
  }
  __syncthreads();
  const float m1   = redS[0] * (1.0f / D_MODEL);
  const float var1 = redQ[0] * (1.0f / D_MODEL) - m1 * m1;
  const float inv1 = __frsqrt_rn(var1 + eps);
  __syncthreads();

  float s2 = 0.0f, q2 = 0.0f;
  for (int n = tid; n < D_MODEL; n += 256) {
    const float y1 = (buf[n] - m1) * inv1 * ln_w[n] + ln_b[n];
    buf[n] = y1; s2 += y1; q2 += y1 * y1;
  }
#pragma unroll
  for (int o = 16; o; o >>= 1) { s2 += __shfl_xor(s2, o, 32); q2 += __shfl_xor(q2, o, 32); }
  if (lane == 0) { redS[wave] = s2; redQ[wave] = q2; }
  __syncthreads();
  if (wave == 0) {
    s2 = (lane < 8) ? redS[lane] : 0.0f;
    q2 = (lane < 8) ? redQ[lane] : 0.0f;
#pragma unroll
    for (int o = 16; o; o >>= 1) { s2 += __shfl_xor(s2, o, 32); q2 += __shfl_xor(q2, o, 32); }
    if (lane == 0) { redS[0] = s2; redQ[0] = q2; }
  }
  __syncthreads();
  const float m2   = redS[0] * (1.0f / D_MODEL);
  const float var2 = redQ[0] * (1.0f / D_MODEL) - m2 * m2;
  const float inv2 = __frsqrt_rn(var2 + eps);

  const float* sc = ss + (long long)b * 2 * D_MODEL;
  for (int n = tid; n < D_MODEL; n += 256) {
    const float h = (buf[n] - m2) * inv2;
    out[(long long)row * D_MODEL + n] = h * (1.0f + sc[n]) + sc[D_MODEL + n];
  }
}

// ---------------------------------------------------------------------------
extern "C" void kernel_launch(void* const* d_in, const int* in_sizes, int n_in,
                              void* d_out, int out_size, void* d_ws, size_t ws_size,
                              hipStream_t stream) {
  const float* x         = (const float*)d_in[0];
  const float* condition = (const float*)d_in[1];
  const float* in_proj_w = (const float*)d_in[2];
  const float* in_proj_b = (const float*)d_in[3];
  const float* conv_w    = (const float*)d_in[4];
  const float* conv_b    = (const float*)d_in[5];
  const float* x_proj_w  = (const float*)d_in[6];
  const float* dt_proj_w = (const float*)d_in[7];
  const float* dt_proj_b = (const float*)d_in[8];
  const float* A_log     = (const float*)d_in[9];
  const float* Dskip     = (const float*)d_in[10];
  const float* out_proj_w= (const float*)d_in[11];
  const float* out_proj_b= (const float*)d_in[12];
  const float* ln_w      = (const float*)d_in[13];
  const float* ln_b      = (const float*)d_in[14];
  const float* ada_w     = (const float*)d_in[15];
  const float* ada_b     = (const float*)d_in[16];

  float* out0      = (float*)d_out;                         // (B,SEQ,D_MODEL)
  float* new_state = out0 + (size_t)NROWS * D_MODEL;        // (B,D_INNER,D_STATE)

  // ---- workspace layout: f32 region, then f16 region (all 32B aligned) ----
  float* ws  = (float*)d_ws;
  float* xz  = ws;                                          // NROWS*3072
  float* xc  = xz  + (size_t)NROWS * 2 * D_INNER;           // NROWS*1536
  float* dbc = xc  + (size_t)NROWS * D_INNER;               // NROWS*176
  float* dt  = dbc + (size_t)NROWS * (DT_RANK + 2*D_STATE); // NROWS*1536
  float* mam = dt  + (size_t)NROWS * D_INNER;               // NROWS*768
  float* ss  = mam + (size_t)NROWS * D_MODEL;               // 4*1536

  _Float16* h16  = (_Float16*)(ss + (size_t)BATCH * 2 * D_MODEL);
  _Float16* xh   = h16;                                     // NROWS*768
  _Float16* wih  = xh   + (size_t)NROWS * D_MODEL;          // 3072*768
  _Float16* xch  = wih  + (size_t)2 * D_INNER * D_MODEL;    // NROWS*1536
  _Float16* wxh  = xch  + (size_t)NROWS * D_INNER;          // 176*1536
  _Float16* dtrh = wxh  + (size_t)(DT_RANK + 2*D_STATE) * D_INNER; // NROWS*64
  _Float16* wdh  = dtrh + (size_t)NROWS * DT_PAD;           // 1536*64
  _Float16* ysh  = wdh  + (size_t)D_INNER * DT_PAD;         // NROWS*1536
  _Float16* woh  = ysh  + (size_t)NROWS * D_INNER;          // 768*1536

  const dim3 blk(256);
  auto cdiv = [](int a, int b) { return (a + b - 1) / b; };
  // a block covers 8 waves * 8 tiles * 16 = 1024 output columns
  auto gridY = [&](int N) { return cdiv(N, 1024); };

  // 0) convert weights + x to f16 (pads dt_proj_w K: 48 -> 64)
  cvt_f16_kernel<<<cdiv(NROWS * D_MODEL, 256), blk, 0, stream>>>(x, xh, NROWS * D_MODEL);
  cvt_f16_kernel<<<cdiv(2 * D_INNER * D_MODEL, 256), blk, 0, stream>>>(in_proj_w, wih, 2 * D_INNER * D_MODEL);
  cvt_f16_kernel<<<cdiv((DT_RANK + 2 * D_STATE) * D_INNER, 256), blk, 0, stream>>>(x_proj_w, wxh, (DT_RANK + 2 * D_STATE) * D_INNER);
  cvt_f16_kernel<<<cdiv(D_MODEL * D_INNER, 256), blk, 0, stream>>>(out_proj_w, woh, D_MODEL * D_INNER);
  pack_pad_f16_kernel<<<cdiv(D_INNER * DT_PAD, 256), blk, 0, stream>>>(
      dt_proj_w, DT_RANK, DT_RANK, wdh, DT_PAD, D_INNER * DT_PAD);

  // 1) xz = x @ in_proj_w^T + b            M=8192 N=3072 K=768
  wmma_gemm_kernel<0><<<dim3(NROWS / 16, gridY(2 * D_INNER)), blk, 0, stream>>>(
      xh, D_MODEL, wih, D_MODEL, in_proj_b, xz, 2 * D_INNER, 2 * D_INNER, D_MODEL);

  // 2) xc = silu(causal_conv(xin))  (f32 + f16 copies)
  conv_silu_kernel<<<cdiv(NROWS * D_INNER, 256), blk, 0, stream>>>(
      xz, conv_w, conv_b, xc, xch, NROWS * D_INNER);

  // 3) dbc = xc @ x_proj_w^T               M=8192 N=176 K=1536 (no bias)
  wmma_gemm_kernel<0><<<dim3(NROWS / 16, gridY(DT_RANK + 2 * D_STATE)), blk, 0, stream>>>(
      xch, D_INNER, wxh, D_INNER, nullptr, dbc, DT_RANK + 2 * D_STATE,
      DT_RANK + 2 * D_STATE, D_INNER);

  // 3b) pack dt-rank slice of dbc, zero-padded to K=64
  pack_pad_f16_kernel<<<cdiv(NROWS * DT_PAD, 256), blk, 0, stream>>>(
      dbc, DT_RANK + 2 * D_STATE, DT_RANK, dtrh, DT_PAD, NROWS * DT_PAD);

  // 4) dt = softplus(dtr @ dt_proj_w^T + b)  M=8192 N=1536 K=64 (padded)
  wmma_gemm_kernel<1><<<dim3(NROWS / 16, gridY(D_INNER)), blk, 0, stream>>>(
      dtrh, DT_PAD, wdh, DT_PAD, dt_proj_b, dt, D_INNER, D_INNER, DT_PAD);

  // 5) selective scan (fused D-skip + silu(z) gate) -> ysh (f16) + new_state
  scan_kernel<<<dim3((BATCH * D_INNER) / 8), blk, 0, stream>>>(
      xc, dt, dbc, xz, A_log, Dskip, ysh, new_state);

  // 6) mam = ys @ out_proj_w^T + b         M=8192 N=768 K=1536
  wmma_gemm_kernel<0><<<dim3(NROWS / 16, gridY(D_MODEL)), blk, 0, stream>>>(
      ysh, D_INNER, woh, D_INNER, out_proj_b, mam, D_MODEL, D_MODEL, D_INNER);

  // 7) ss = condition @ ada_w^T + ada_b
  ada_kernel<<<cdiv(BATCH * 2 * D_MODEL, 256), blk, 0, stream>>>(
      condition, ada_w, ada_b, ss);

  // 8) fused double layernorm + AdaLN modulation
  final_ln_kernel<<<dim3(NROWS), blk, 0, stream>>>(x, mam, ln_w, ln_b, ss, out0);
}
